// LSTM_38732015075686
// MI455X (gfx1250) — matrix operand order repaired and checked by reading, not compiled, not run on hardware
//
#include <hip/hip_runtime.h>
#include <math.h>

// ---------------- problem constants (from reference) ----------------
#define NB   2048          // batch
#define DD   16            // obs dim
#define EE   256           // embed dim
#define HH   768           // hidden
#define GG   2304          // 3*H
#define TT   8             // observed timesteps
#define NPRED 12           // n_predict (python scalar, known constant)
#define EMB_SCALE 4.0f

typedef __attribute__((ext_vector_type(16))) _Float16 v16h;
typedef __attribute__((ext_vector_type(8)))  _Float16 v8h;
typedef __attribute__((ext_vector_type(8)))  float    v8f;

// ---------------- small elementwise kernels ----------------

__global__ void k_f32_to_f16(const float* __restrict__ s, _Float16* __restrict__ d, int n) {
    int i = blockIdx.x * 256 + threadIdx.x;
    if (i < n) d[i] = (_Float16)s[i];
}

__global__ void k_zero_u32(unsigned int* __restrict__ p, int n) {
    int i = blockIdx.x * 256 + threadIdx.x;
    if (i < n) p[i] = 0u;
}

// x[n, 0:256] = EMB_SCALE * (s2[n]-s1[n]) @ embW^T + embb   (f16 out, row stride ldx)
__global__ void k_embed(const float* __restrict__ s2, const float* __restrict__ s1,
                        const float* __restrict__ W, const float* __restrict__ b,
                        _Float16* __restrict__ x, int ldx) {
    __shared__ float v[DD];
    int n = blockIdx.x;
    int e = threadIdx.x;               // 256 threads == EE
    if (e < DD) v[e] = s2[(size_t)n * DD + e] - s1[(size_t)n * DD + e];
    __syncthreads();
    float acc = b[e];
    #pragma unroll
    for (int d = 0; d < DD; ++d)
        acc = fmaf(EMB_SCALE * v[d], W[e * DD + d], acc);
    x[(size_t)n * ldx + e] = (_Float16)acc;
}

// x2[n, 256:1024] = h_inv (f16); written once, invariant over encoder-2 scan
__global__ void k_copy_hinv(const _Float16* __restrict__ hb, _Float16* __restrict__ x2) {
    int idx = blockIdx.x * 256 + threadIdx.x;
    if (idx >= NB * HH) return;
    int n = idx / HH, j = idx - n * HH;
    x2[(size_t)n * 1024 + EE + j] = hb[idx];
}

// ---------------- WMMA fragment loaders (ISA 7.12.2 layouts) ----------------
// A fragment (16-bit A 16x32): lane L (r=L&15, half=L>>4) holds
//   A[r][half*8 + 0..7] then A[r][16 + half*8 + 0..7]  -> two 16B loads.
// B fragment (32x16 K-major): lane holds W[col][half*8 + 0..7] then W[col][16 + half*8 + 0..7].

__device__ inline v16h load_frag_a(const _Float16* __restrict__ A, int lda,
                                   int row0, int k0, int lane) {
    int r = lane & 15, half = lane >> 4;
    const _Float16* p = A + (size_t)(row0 + r) * lda + k0 + half * 8;
    v8h lo = *reinterpret_cast<const v8h*>(p);
    v8h hi = *reinterpret_cast<const v8h*>(p + 16);
    v16h out;
    #pragma unroll
    for (int i = 0; i < 8; ++i) { out[i] = lo[i]; out[i + 8] = hi[i]; }
    return out;
}

__device__ inline v16h load_frag_b(const _Float16* __restrict__ W, int ldb,
                                   int col0, int k0, int lane) {
    int c = lane & 15, half = lane >> 4;
    const _Float16* p = W + (size_t)(col0 + c) * ldb + k0 + half * 8;
    v8h lo = *reinterpret_cast<const v8h*>(p);
    v8h hi = *reinterpret_cast<const v8h*>(p + 16);
    v16h out;
    #pragma unroll
    for (int i = 0; i < 8; ++i) { out[i] = lo[i]; out[i + 8] = hi[i]; }
    return out;
}

#define WMMA_F16(a, b, c) \
    __builtin_amdgcn_wmma_f32_16x16x32_f16(false, (a), false, (b), (short)0, (c), false, false)

// ---------------- fully fused GRU step ----------------
// One launch computes h_new = GRU(x, h_old) with NO gi/gh global buffers.
// Block = 8 waves: wm in {0,1} (32-row halves of a 64-row patch),
//                  wj in {0..3} (16-j strips of a 64-j patch).
// Each wave accumulates 6 gate tiles (r,z,n) x (input, hidden) of 32x16 in v8f regs;
// the WMMA C layout puts the SAME (row, jcol) in the same lane/element across all
// six accumulators, so the gate nonlinearity fuses elementwise in the epilogue.
// All 10 fragment loads are hoisted ahead of the 6 WMMAs so they issue as one
// clause and drain with staged partial s_wait_loadcnt under WMMA execution.

__global__ __launch_bounds__(256)
void k_gru_fused(const _Float16* __restrict__ Ax, int ldax, int Kx,
                 const _Float16* __restrict__ Wih,    // [2304 x Kx], rows = [r|z|n] blocks
                 const float* __restrict__ bih,
                 const _Float16* __restrict__ Ah,     // [2048 x 768] old h (f16)
                 const _Float16* __restrict__ Whh,    // [2304 x 768]
                 const float* __restrict__ bhh,
                 const float* __restrict__ h_old,     // [2048 x 768] f32
                 float* __restrict__ h_new,
                 _Float16* __restrict__ hb_new) {
    int lane   = threadIdx.x & 31;
    int waveId = threadIdx.x >> 5;
    int wm = waveId >> 2;                        // 0..1
    int wj = waveId & 3;                         // 0..3
    int rowBase = blockIdx.x * 64 + wm * 32;
    int jBase   = blockIdx.y * 64 + wj * 16;

    v8f ci[3][2] = {};   // input-path gates  [r,z,n][m-tile]
    v8f ch[3][2] = {};   // hidden-path gates [r,z,n][m-tile]

    // ---- input path: gi = x @ Wih^T (columns jBase + g*768) ----
    for (int k0 = 0; k0 < Kx; k0 += 32) {
        if (k0 + 32 < Kx) {
            __builtin_prefetch(Ax  + (size_t)(rowBase + (lane & 15)) * ldax + k0 + 32, 0, 1);
            __builtin_prefetch(Wih + (size_t)(jBase   + (lane & 15)) * Kx   + k0 + 32, 0, 1);
        }
        v16h a0 = load_frag_a(Ax, ldax, rowBase,      k0, lane);
        v16h a1 = load_frag_a(Ax, ldax, rowBase + 16, k0, lane);
        v16h b0 = load_frag_b(Wih, Kx,          jBase, k0, lane);
        v16h b1 = load_frag_b(Wih, Kx, HH     + jBase, k0, lane);
        v16h b2 = load_frag_b(Wih, Kx, 2 * HH + jBase, k0, lane);
        ci[0][0] = WMMA_F16(a0, b0, ci[0][0]);
        ci[0][1] = WMMA_F16(a1, b0, ci[0][1]);
        ci[1][0] = WMMA_F16(a0, b1, ci[1][0]);
        ci[1][1] = WMMA_F16(a1, b1, ci[1][1]);
        ci[2][0] = WMMA_F16(a0, b2, ci[2][0]);
        ci[2][1] = WMMA_F16(a1, b2, ci[2][1]);
    }
    // ---- hidden path: gh = h_old @ Whh^T ----
    for (int k0 = 0; k0 < HH; k0 += 32) {
        if (k0 + 32 < HH) {
            __builtin_prefetch(Ah  + (size_t)(rowBase + (lane & 15)) * HH + k0 + 32, 0, 1);
            __builtin_prefetch(Whh + (size_t)(jBase   + (lane & 15)) * HH + k0 + 32, 0, 1);
        }
        v16h a0 = load_frag_a(Ah, HH, rowBase,      k0, lane);
        v16h a1 = load_frag_a(Ah, HH, rowBase + 16, k0, lane);
        v16h b0 = load_frag_b(Whh, HH,          jBase, k0, lane);
        v16h b1 = load_frag_b(Whh, HH, HH     + jBase, k0, lane);
        v16h b2 = load_frag_b(Whh, HH, 2 * HH + jBase, k0, lane);
        ch[0][0] = WMMA_F16(a0, b0, ch[0][0]);
        ch[0][1] = WMMA_F16(a1, b0, ch[0][1]);
        ch[1][0] = WMMA_F16(a0, b1, ch[1][0]);
        ch[1][1] = WMMA_F16(a1, b1, ch[1][1]);
        ch[2][0] = WMMA_F16(a0, b2, ch[2][0]);
        ch[2][1] = WMMA_F16(a1, b2, ch[2][1]);
    }

    // ---- fused GRU gate epilogue ----
    // C/D layout: lane L, elem e -> row = (L>>4)*8 + e, col = L&15 (per 16x16 tile);
    // col is constant per lane, so gate biases load once.
    int nl = lane & 15, half = lane >> 4;
    int jcol = jBase + nl;
    float bir = bih[jcol],          bhr = bhh[jcol];
    float biz = bih[HH + jcol],     bhz = bhh[HH + jcol];
    float bin = bih[2 * HH + jcol], bhn = bhh[2 * HH + jcol];
    #pragma unroll
    for (int i = 0; i < 2; ++i) {
        #pragma unroll
        for (int e = 0; e < 8; ++e) {
            int row = rowBase + i * 16 + half * 8 + e;
            float r  = 1.f / (1.f + __expf(-(ci[0][i][e] + bir + ch[0][i][e] + bhr)));
            float z  = 1.f / (1.f + __expf(-(ci[1][i][e] + biz + ch[1][i][e] + bhz)));
            float nn = tanhf(ci[2][i][e] + bin + r * (ch[2][i][e] + bhn));
            float hv = h_old[(size_t)row * HH + jcol];
            float o  = (1.f - z) * nn + z * hv;
            h_new[(size_t)row * HH + jcol]  = o;
            hb_new[(size_t)row * HH + jcol] = (_Float16)o;
        }
    }
}

// ---------------- decoder head: h2n transform + mix + relu residual ----------------
__device__ inline float softplusf(float x) { return (x > 20.f) ? x : log1pf(__expf(x)); }

__global__ void k_head(const float* __restrict__ h,
                       const float* __restrict__ W5, const float* __restrict__ b5,
                       const float* __restrict__ Wm, const float* __restrict__ bm,
                       const float* __restrict__ obs2,
                       float* __restrict__ rel, float* __restrict__ pos) {
    int wave = threadIdx.x >> 5, lane = threadIdx.x & 31;
    int n = blockIdx.x * 8 + wave;                 // 256 blocks * 8 waves = 2048 rows
    float s[5] = {0.f, 0.f, 0.f, 0.f, 0.f};
    const float* hr = h + (size_t)n * HH;
    for (int k = lane; k < HH; k += 32) {
        float hv = hr[k];
        #pragma unroll
        for (int j = 0; j < 5; ++j) s[j] = fmaf(hv, W5[j * HH + k], s[j]);
    }
    #pragma unroll
    for (int j = 0; j < 5; ++j)
        #pragma unroll
        for (int off = 16; off; off >>= 1) s[j] += __shfl_xor(s[j], off, 32);

    float t[5];
    t[0] = s[0] + b5[0];
    t[1] = s[1] + b5[1];
    t[2] = 0.01f + 0.2f * softplusf(s[2] + b5[2]);
    t[3] = 0.01f + 0.2f * softplusf(s[3] + b5[3]);
    t[4] = 0.7f * tanhf(s[4] + b5[4]);

    if (lane == 0) {
        #pragma unroll
        for (int j = 0; j < 5; ++j) rel[(size_t)n * 5 + j] = t[j];
    }
    if (lane < DD) {
        int d = lane;
        float m = bm[d];
        #pragma unroll
        for (int j = 0; j < 5; ++j) m = fmaf(t[j], Wm[d * 5 + j], m);
        pos[(size_t)n * DD + d] = obs2[(size_t)n * DD + d] + fmaxf(m, 0.f);
    }
}

// ---------------- host orchestration ----------------

extern "C" void kernel_launch(void* const* d_in, const int* in_sizes, int n_in,
                              void* d_out, int out_size, void* d_ws, size_t ws_size,
                              hipStream_t stream) {
    (void)in_sizes; (void)n_in; (void)out_size; (void)ws_size;

    const float* obs    = (const float*)d_in[0];
    const float* embW   = (const float*)d_in[1];
    const float* embb   = (const float*)d_in[2];
    const float* e1Wih  = (const float*)d_in[3];
    const float* e1Whh  = (const float*)d_in[4];
    const float* e1bih  = (const float*)d_in[5];
    const float* e1bhh  = (const float*)d_in[6];
    const float* e2Wih  = (const float*)d_in[7];
    const float* e2Whh  = (const float*)d_in[8];
    const float* e2bih  = (const float*)d_in[9];
    const float* e2bhh  = (const float*)d_in[10];
    const float* dcWih  = (const float*)d_in[11];
    const float* dcWhh  = (const float*)d_in[12];
    const float* dcbih  = (const float*)d_in[13];
    const float* dcbhh  = (const float*)d_in[14];
    const float* h2nW   = (const float*)d_in[15];
    const float* h2nb   = (const float*)d_in[16];
    const float* mixW   = (const float*)d_in[17];
    const float* mixb   = (const float*)d_in[18];

    float* out = (float*)d_out;
    float* rel_base  = out;                                // (12, N, 5)
    float* pred_base = out + (size_t)NPRED * NB * 5;       // (12, N, 16) -- also feedback buf

    // workspace carve-up
    char* base = (char*)d_ws;
    size_t cur = 0;
    auto alloc = [&](size_t bytes) -> char* {
        char* p = base + cur;
        cur = (cur + bytes + 255) & ~(size_t)255;
        return p;
    };
    _Float16* w_e1i = (_Float16*)alloc((size_t)GG * EE   * 2);
    _Float16* w_e1h = (_Float16*)alloc((size_t)GG * HH   * 2);
    _Float16* w_e2i = (_Float16*)alloc((size_t)GG * 1024 * 2);
    _Float16* w_e2h = (_Float16*)alloc((size_t)GG * HH   * 2);
    _Float16* w_dci = (_Float16*)alloc((size_t)GG * EE   * 2);
    _Float16* w_dch = (_Float16*)alloc((size_t)GG * HH   * 2);
    _Float16* x2    = (_Float16*)alloc((size_t)NB * 1024 * 2);   // [emb | h_inv]
    _Float16* hbA   = (_Float16*)alloc((size_t)NB * HH   * 2);   // h ping-pong (f16)
    _Float16* hbB   = (_Float16*)alloc((size_t)NB * HH   * 2);
    float*    hfA   = (float*)   alloc((size_t)NB * HH   * 4);   // h ping-pong (f32)
    float*    hfB   = (float*)   alloc((size_t)NB * HH   * 4);

    auto conv = [&](const float* s, _Float16* d, int n) {
        k_f32_to_f16<<<(n + 255) / 256, 256, 0, stream>>>(s, d, n);
    };
    conv(e1Wih, w_e1i, GG * EE);
    conv(e1Whh, w_e1h, GG * HH);
    conv(e2Wih, w_e2i, GG * 1024);
    conv(e2Whh, w_e2h, GG * HH);
    conv(dcWih, w_dci, GG * EE);
    conv(dcWhh, w_dch, GG * HH);

    // double-buffered h state
    float*    cur_hf = hfA; float*    nxt_hf = hfB;
    _Float16* cur_hb = hbA; _Float16* nxt_hb = hbB;

    auto zero_cur_h = [&]() {
        int nf = NB * HH;          // f32 words
        int nh = NB * HH / 2;      // f16 pairs as u32
        k_zero_u32<<<(nf + 255) / 256, 256, 0, stream>>>((unsigned int*)cur_hf, nf);
        k_zero_u32<<<(nh + 255) / 256, 256, 0, stream>>>((unsigned int*)cur_hb, nh);
    };

    dim3 fgrid(NB / 64, HH / 64);  // (32, 12)
    auto gru_step = [&](const _Float16* Ax, int ldax, int Kx,
                        const _Float16* Wi, const float* bi,
                        const _Float16* Wh, const float* bh) {
        k_gru_fused<<<fgrid, 256, 0, stream>>>(Ax, ldax, Kx, Wi, bi,
                                               cur_hb, Wh, bh,
                                               cur_hf, nxt_hf, nxt_hb);
        float*    tf = cur_hf; cur_hf = nxt_hf; nxt_hf = tf;
        _Float16* tb = cur_hb; cur_hb = nxt_hb; nxt_hb = tb;
    };
    auto embed = [&](const float* s2, const float* s1) {
        k_embed<<<NB, 256, 0, stream>>>(s2, s1, embW, embb, x2, 1024);
    };

    // ---- encoder 1 (backward velocities): vel_b[t] = obs[7-t] - obs[6-t] ----
    zero_cur_h();
    for (int t = 0; t < TT - 1; ++t) {
        const float* s2 = obs + (size_t)(TT - 1 - t) * NB * DD;
        const float* s1 = obs + (size_t)(TT - 2 - t) * NB * DD;
        embed(s2, s1);
        gru_step(x2, 1024, EE, w_e1i, e1bih, w_e1h, e1bhh);
    }
    // freeze h_inv into x2 columns 256..1023 (invariant over encoder-2 scan)
    k_copy_hinv<<<(NB * HH + 255) / 256, 256, 0, stream>>>(cur_hb, x2);

    // ---- encoder 2 (forward velocities): vel_f[t] = obs[t+1] - obs[t] ----
    zero_cur_h();
    for (int t = 0; t < TT - 1; ++t) {
        const float* s1 = obs + (size_t)t * NB * DD;
        const float* s2 = obs + (size_t)(t + 1) * NB * DD;
        embed(s2, s1);
        gru_step(x2, 1024, 1024, w_e2i, e2bih, w_e2h, e2bhh);
    }

    // ---- decoder: 12 autoregressive steps, feedback through pred region of d_out ----
    for (int t = 0; t < NPRED; ++t) {
        const float *s1, *s2;
        if (t == 0)      { s1 = obs + (size_t)(TT - 2) * NB * DD; s2 = obs + (size_t)(TT - 1) * NB * DD; }
        else if (t == 1) { s1 = obs + (size_t)(TT - 1) * NB * DD; s2 = pred_base; }
        else             { s1 = pred_base + (size_t)(t - 2) * NB * DD;
                           s2 = pred_base + (size_t)(t - 1) * NB * DD; }
        embed(s2, s1);
        gru_step(x2, 1024, EE, w_dci, dcbih, w_dch, dcbhh);
        k_head<<<NB / 8, 256, 0, stream>>>(cur_hf, h2nW, h2nb, mixW, mixb, s2,
                                           rel_base + (size_t)t * NB * 5,
                                           pred_base + (size_t)t * NB * DD);
    }
}
